// DeepSeekQNN_43439299231938
// MI455X (gfx1250) — matrix operand order repaired and checked by reading, not compiled, not run on hardware
//
#include <hip/hip_runtime.h>
#include <hip/hip_bf16.h>
#include <math.h>

// ---------------- problem dims (compile-time) ----------------
#define NLAYER 2
#define NEXP   32
#define DMODEL 512
#define HID    2048
#define VOCAB  50257
#define TSEQ   2048
#define CAP    128     // expert capacity = 2*T/E
#define KTOP   32

// ---------------- CDNA5 WMMA types ----------------
typedef __attribute__((ext_vector_type(16))) __bf16 v16bf;
typedef __attribute__((ext_vector_type(8)))  float  v8f;

union Frag16 { v16bf v; unsigned short u[16]; unsigned int w[8]; };

// round-to-nearest-even f32 -> bf16 (bit trick)
__device__ __forceinline__ unsigned short f2bf(float f) {
  unsigned int u = __builtin_bit_cast(unsigned int, f);
  u += 0x7FFFu + ((u >> 16) & 1u);
  return (unsigned short)(u >> 16);
}
__device__ __forceinline__ unsigned int pack2bf(float x, float y) {
  return (unsigned int)f2bf(x) | ((unsigned int)f2bf(y) << 16);
}

// =============================================================
// Generic WMMA GEMM: D = epi(A*B*scale + bias)
//   A: f32 [M x K] (optionally row-gathered), B: f32 [K x N] ([N x K] if BT).
//   B tile is cooperatively staged to LDS (f32 -> bf16 at store) in a
//   transposed, padded [n][k] layout so each lane's fragment is two runs of 4
//   contiguous dwords of K-pairs.  A is loaded directly (4 x b128/lane).
//   One 16x16 tile per wave32, 8 waves/block -> 32x64 block tile.
//   Out-of-range rows/cols are CLAMPED (not guarded): garbage only ever
//   lands in rows/cols that the store guard discards, so there is zero
//   EXEC divergence in the hot loop (WMMA requires EXEC == all ones).
// =============================================================
template<bool GATHER, bool BT, bool BIAS, bool GELU, bool SCATTER>
__global__ void __launch_bounds__(256)
k_wmma_gemm(const float* __restrict__ A, int lda, long long strideA,
            const int* __restrict__ rowIdx, int idxStride,
            const float* __restrict__ B, int ldb, long long strideB,
            const float* __restrict__ bias, int strideBias,
            float* __restrict__ Dst, int ldd, long long strideD,
            const int* __restrict__ outIdx, const float* __restrict__ rowScale,
            int M, int N, int Kdim, float scale)
{
  // bf16 B tile [64 n][32 k], row stride padded to 34 (17 dwords) vs 32
  // to break the 16-bank stride pattern on fragment reads.
  __shared__ unsigned short Bs[64 * 34];

  const int tid  = threadIdx.x;
  const int lane = tid & 31;
  const int wave = tid >> 5;
  const int half = lane >> 4;      // lane group (ISA 16-bit A/B layout)
  const int l16  = lane & 15;
  const int tileM  = blockIdx.y * 32 + (wave >> 2) * 16;
  const int tileNb = blockIdx.x * 64;              // block N origin
  const int z = blockIdx.z;

  // ---- A row pointer for this lane (clamped / gathered, branch-free) ----
  const int mg  = tileM + l16;
  const int mgc = (mg < M) ? mg : (M - 1);
  const float* Ab = A + (long long)z * strideA;
  long long arowOff;
  if (GATHER) {
    arowOff = (long long)rowIdx[(long long)z * idxStride + mgc] * lda;
  } else {
    arowOff = (long long)mgc * lda;
  }
  const float* Ap = Ab + arowOff;

  const float* Bb = B + (long long)z * strideB;
  const unsigned int* Bw = (const unsigned int*)Bs;
  const int bbase = ((wave & 3) * 16 + l16) * 17 + 4 * half;

  v8f acc = {};
  for (int kb = 0; kb < Kdim; kb += 32) {
    // ---------------- stage B tile into LDS ----------------
    __syncthreads();   // previous iteration's fragment reads complete
    if (BT) {
      // B stored [N x K]: thread -> (n row r, k chunk kc), contiguous in K
      const int r  = tid >> 2;                 // 0..63
      const int kc = (tid & 3) * 8;            // 0,8,16,24
      const int nr = tileNb + r;
      const int nrc = (nr < N) ? nr : (N - 1);
      const float* src = Bb + (long long)nrc * ldb + kb + kc;
      __builtin_prefetch(src + 32, 0, 0);      // next K tile
      const float4 b0 = *(const float4*)(src);
      const float4 b1 = *(const float4*)(src + 4);
      unsigned int* Bwst = (unsigned int*)Bs;
      const int o = r * 17 + (kc >> 1);
      Bwst[o + 0] = pack2bf(b0.x, b0.y);
      Bwst[o + 1] = pack2bf(b0.z, b0.w);
      Bwst[o + 2] = pack2bf(b1.x, b1.y);
      Bwst[o + 3] = pack2bf(b1.z, b1.w);
    } else {
      // B stored [K x N]: thread -> (k row r, 8 consecutive n)
      const int r  = tid >> 3;                 // 0..31
      const int cb = (tid & 7) * 8;            // 0..56
      const float* Brow = Bb + (long long)(kb + r) * ldb;
      __builtin_prefetch(Brow + 32LL * ldb + tileNb + cb, 0, 0); // next K tile
      if (tileNb + 64 <= N) {
        const float4 b0 = *(const float4*)(Brow + tileNb + cb);
        const float4 b1 = *(const float4*)(Brow + tileNb + cb + 4);
        const float vals[8] = {b0.x, b0.y, b0.z, b0.w, b1.x, b1.y, b1.z, b1.w};
#pragma unroll
        for (int i = 0; i < 8; ++i) Bs[(cb + i) * 34 + r] = f2bf(vals[i]);
      } else {                                  // ragged right edge (uniform branch)
#pragma unroll
        for (int i = 0; i < 8; ++i) {
          int n = tileNb + cb + i; if (n >= N) n = N - 1;
          Bs[(cb + i) * 34 + r] = f2bf(Brow[n]);
        }
      }
    }
    __syncthreads();

    // ---------------- A fragment: 4 x b128 per lane ----------------
    // lane covers K chunks [8h, 8h+8) and [16+8h, 16+8h+8)
    __builtin_prefetch(Ap + kb + 32, 0, 0);
    const float4 a0 = *(const float4*)(Ap + kb + 8 * half);
    const float4 a1 = *(const float4*)(Ap + kb + 8 * half + 4);
    const float4 a2 = *(const float4*)(Ap + kb + 8 * half + 16);
    const float4 a3 = *(const float4*)(Ap + kb + 8 * half + 20);
    Frag16 fa, fb;
    fa.w[0] = pack2bf(a0.x, a0.y); fa.w[1] = pack2bf(a0.z, a0.w);
    fa.w[2] = pack2bf(a1.x, a1.y); fa.w[3] = pack2bf(a1.z, a1.w);
    fa.w[4] = pack2bf(a2.x, a2.y); fa.w[5] = pack2bf(a2.z, a2.w);
    fa.w[6] = pack2bf(a3.x, a3.y); fa.w[7] = pack2bf(a3.z, a3.w);

    // ---------------- B fragment: 2 x 4 contiguous dwords from LDS ----------------
#pragma unroll
    for (int j = 0; j < 8; ++j) fb.w[j] = Bw[bbase + j + (j >= 4 ? 4 : 0)];

    acc = __builtin_amdgcn_wmma_f32_16x16x32_bf16(false, fa.v, false, fb.v,
                                                  (short)0, acc, false, false);
  }

  // C/D 16x16 f32 layout: lane<16 -> M=r, lane>=16 -> M=8+r; N = lane&15
  const int ng = tileNb + (wave & 3) * 16 + l16;
  if (ng < N) {
    float bv = 0.f;
    if (BIAS) bv = bias[(long long)z * strideBias + ng];
#pragma unroll
    for (int r = 0; r < 8; ++r) {
      const int mo = tileM + r + 8 * half;
      if (mo < M) {
        float vv = acc[r] * scale + bv;
        if (GELU) vv = 0.5f * vv * (1.f + erff(vv * 0.70710678118654752f));
        if (SCATTER) {
          const int   orow = outIdx[(long long)z * idxStride + mo];
          const float gs   = rowScale[(long long)z * idxStride + mo];
          atomicAdd(Dst + (long long)orow * ldd + ng, vv * gs);
        } else {
          float* Db = Dst + (long long)z * strideD;
          Db[(long long)mo * ldd + ng] = vv;
        }
      }
    }
  }
}

// ---------------- small helper kernels ----------------
__global__ void k_init_loss(float* loss) { if (threadIdx.x == 0) *loss = 0.f; }

__global__ void __launch_bounds__(256) k_embed(const int* __restrict__ idx,
                                               const float* __restrict__ tok,
                                               const float* __restrict__ pos,
                                               float* __restrict__ x)
{
  const int t = blockIdx.x;
  const long long trow = (long long)idx[t] * DMODEL;
  for (int d = threadIdx.x; d < DMODEL; d += 256)
    x[(long long)t*DMODEL + d] = tok[trow + d] + pos[(long long)t*DMODEL + d];
}

// q,k scaled elementwise by interleaved sin/cos positional encoding (faithful)
__global__ void __launch_bounds__(256) k_rotary(const float* __restrict__ q,
                                                const float* __restrict__ k,
                                                float* __restrict__ qr,
                                                float* __restrict__ kr)
{
  const int t = blockIdx.x;
  const float c = -logf(10000.f) / (float)DMODEL;
  for (int d = threadIdx.x; d < DMODEL; d += 256) {
    const int i2 = d & ~1;
    const float ang = (float)t * expf((float)i2 * c);
    const float pe = (d & 1) ? cosf(ang) : sinf(ang);
    const long long o = (long long)t*DMODEL + d;
    qr[o] = q[o] * pe;
    kr[o] = k[o] * pe;
  }
}

// per-row: threshold at 32nd-largest (counting multiplicity), zeros elsewhere,
// softmax over the FULL row (zeros contribute exp(0)); in-place on scores.
__global__ void __launch_bounds__(256) k_topk_softmax(float* __restrict__ scores)
{
  __shared__ float row[TSEQ];
  __shared__ float rv[256];
  __shared__ int   ri[256];
  __shared__ float kth_s;
  const int r = blockIdx.x;
  float* S = scores + (long long)r * TSEQ;
  for (int i = threadIdx.x; i < TSEQ; i += 256) row[i] = S[i];
  __syncthreads();
  for (int it = 0; it < KTOP; ++it) {
    float best = -INFINITY; int bi = 0;
    for (int i = threadIdx.x; i < TSEQ; i += 256) {
      const float vv = row[i];
      if (vv > best) { best = vv; bi = i; }
    }
    rv[threadIdx.x] = best; ri[threadIdx.x] = bi;
    __syncthreads();
    for (int st = 128; st > 0; st >>= 1) {
      if (threadIdx.x < st && rv[threadIdx.x + st] > rv[threadIdx.x]) {
        rv[threadIdx.x] = rv[threadIdx.x + st];
        ri[threadIdx.x] = ri[threadIdx.x + st];
      }
      __syncthreads();
    }
    if (threadIdx.x == 0) { kth_s = rv[0]; row[ri[0]] = -INFINITY; }
    __syncthreads();
  }
  const float kth = kth_s;
  float lm = -INFINITY;
  for (int i = threadIdx.x; i < TSEQ; i += 256) {
    float vv = S[i];
    vv = (vv >= kth) ? vv : 0.f;
    row[i] = vv;
    lm = fmaxf(lm, vv);
  }
  rv[threadIdx.x] = lm; __syncthreads();
  for (int st = 128; st > 0; st >>= 1) {
    if (threadIdx.x < st) rv[threadIdx.x] = fmaxf(rv[threadIdx.x], rv[threadIdx.x + st]);
    __syncthreads();
  }
  const float mx = rv[0]; __syncthreads();
  float ls = 0.f;
  for (int i = threadIdx.x; i < TSEQ; i += 256) {
    const float e = expf(row[i] - mx);
    row[i] = e; ls += e;
  }
  rv[threadIdx.x] = ls; __syncthreads();
  for (int st = 128; st > 0; st >>= 1) {
    if (threadIdx.x < st) rv[threadIdx.x] += rv[threadIdx.x + st];
    __syncthreads();
  }
  const float inv = 1.f / rv[0];
  for (int i = threadIdx.x; i < TSEQ; i += 256) S[i] = row[i] * inv;
}

__global__ void k_gate_softmax(const float* __restrict__ gl, float* __restrict__ pr)
{
  __shared__ float tmp[NEXP];
  const int t = blockIdx.x, e = threadIdx.x;   // blockDim = 32
  const float v = gl[(long long)t*NEXP + e];
  tmp[e] = v; __syncthreads();
  float m = -INFINITY;
  for (int i = 0; i < NEXP; ++i) m = fmaxf(m, tmp[i]);
  __syncthreads();
  const float ex = expf(v - m);
  tmp[e] = ex; __syncthreads();
  float s = 0.f;
  for (int i = 0; i < NEXP; ++i) s += tmp[i];
  pr[(long long)t*NEXP + e] = ex / s;
}

// expert-choice: each expert (block) picks its top-CAP tokens by gate prob;
// also emits per-expert mean gate prob for the aux loss.
__global__ void __launch_bounds__(256) k_expert_topc(const float* __restrict__ probs,
                                                     float* __restrict__ g,
                                                     int* __restrict__ eidx,
                                                     float* __restrict__ mp)
{
  __shared__ float col[TSEQ];
  __shared__ float rv[256];
  __shared__ int   ri[256];
  const int e = blockIdx.x;
  float s = 0.f;
  for (int t = threadIdx.x; t < TSEQ; t += 256) {
    const float p = probs[(long long)t*NEXP + e];
    col[t] = p; s += p;
  }
  rv[threadIdx.x] = s; __syncthreads();
  for (int st = 128; st > 0; st >>= 1) {
    if (threadIdx.x < st) rv[threadIdx.x] += rv[threadIdx.x + st];
    __syncthreads();
  }
  if (threadIdx.x == 0) mp[e] = rv[0] / (float)TSEQ;
  __syncthreads();
  for (int c = 0; c < CAP; ++c) {
    float best = -INFINITY; int bi = 0;
    for (int t = threadIdx.x; t < TSEQ; t += 256) {
      const float pv = col[t];
      if (pv > best) { best = pv; bi = t; }
    }
    rv[threadIdx.x] = best; ri[threadIdx.x] = bi;
    __syncthreads();
    for (int st = 128; st > 0; st >>= 1) {
      if (threadIdx.x < st && rv[threadIdx.x + st] > rv[threadIdx.x]) {
        rv[threadIdx.x] = rv[threadIdx.x + st];
        ri[threadIdx.x] = ri[threadIdx.x + st];
      }
      __syncthreads();
    }
    if (threadIdx.x == 0) {
      g[e*CAP + c] = rv[0];
      eidx[e*CAP + c] = ri[0];
      col[ri[0]] = -INFINITY;
    }
    __syncthreads();
  }
}

// aux = E * mean_e( mean_s(probs) * mean_s(mask) ), mean_s(mask)=CAP/TSEQ exactly.
// The KL load-balance term is EXACTLY zero: expert-choice gives counts = CAP
// per expert per layer, so p == uniform and kl(uniform||p) == 0.
__global__ void k_aux(const float* __restrict__ mp, float* loss)
{
  if (threadIdx.x == 0) {
    float s = 0.f;
    for (int e = 0; e < NEXP; ++e) s += mp[e];
    *loss += s * ((float)CAP / (float)TSEQ);
  }
}

__global__ void __launch_bounds__(256) k_layernorm(const float* __restrict__ x,
                                                   const float* __restrict__ gw,
                                                   const float* __restrict__ bw,
                                                   float* __restrict__ o)
{
  __shared__ float red[256];
  const int t = blockIdx.x;
  const float* xr = x + (long long)t * DMODEL;
  float s = 0.f;
  for (int d = threadIdx.x; d < DMODEL; d += 256) s += xr[d];
  red[threadIdx.x] = s; __syncthreads();
  for (int st = 128; st > 0; st >>= 1) {
    if (threadIdx.x < st) red[threadIdx.x] += red[threadIdx.x + st];
    __syncthreads();
  }
  const float mu = red[0] / (float)DMODEL; __syncthreads();
  float s2 = 0.f;
  for (int d = threadIdx.x; d < DMODEL; d += 256) {
    const float dv = xr[d] - mu; s2 += dv * dv;
  }
  red[threadIdx.x] = s2; __syncthreads();
  for (int st = 128; st > 0; st >>= 1) {
    if (threadIdx.x < st) red[threadIdx.x] += red[threadIdx.x + st];
    __syncthreads();
  }
  const float rstd = rsqrtf(red[0] / (float)DMODEL + 1e-5f);
  for (int d = threadIdx.x; d < DMODEL; d += 256)
    o[(long long)t*DMODEL + d] = (xr[d] - mu) * rstd * gw[d] + bw[d];
}

// ================= host orchestration =================
extern "C" void kernel_launch(void* const* d_in, const int* in_sizes, int n_in,
                              void* d_out, int out_size, void* d_ws, size_t ws_size,
                              hipStream_t stream)
{
  (void)in_sizes; (void)n_in; (void)out_size; (void)ws_size;

  const int*   idx   = (const int*)  d_in[0];
  const float* tok   = (const float*)d_in[1];
  const float* pos   = (const float*)d_in[2];
  const float* Wq    = (const float*)d_in[3];
  const float* bq    = (const float*)d_in[4];
  const float* Wk    = (const float*)d_in[5];
  const float* bk    = (const float*)d_in[6];
  const float* Wv    = (const float*)d_in[7];
  const float* bv    = (const float*)d_in[8];
  const float* gateW = (const float*)d_in[9];
  const float* W1    = (const float*)d_in[10];
  const float* b1    = (const float*)d_in[11];
  const float* W2    = (const float*)d_in[12];
  const float* b2    = (const float*)d_in[13];
  const float* lng   = (const float*)d_in[14];
  const float* lnb   = (const float*)d_in[15];
  const float* headW = (const float*)d_in[16];

  float* logits = (float*)d_out;
  float* loss   = logits + (long long)TSEQ * VOCAB;

  // workspace carve-up (floats)
  float* w = (float*)d_ws;
  const long long TD = (long long)TSEQ * DMODEL;
  float* x   = w; w += TD;
  float* q   = w; w += TD;
  float* kk  = w; w += TD;
  float* v   = w; w += TD;
  float* qr  = w; w += TD;
  float* kr  = w; w += TD;
  float* sc  = w; w += (long long)TSEQ * TSEQ;   // scores -> softmax weights (in place)
  float* xf  = w; w += TD;                       // token stream (residual accumulates here)
  float* gl  = w; w += (long long)TSEQ * NEXP;
  float* pr  = w; w += (long long)TSEQ * NEXP;
  float* g   = w; w += (long long)NEXP * CAP;
  float* mp  = w; w += NEXP;
  float* h   = w; w += (long long)NEXP * CAP * HID;
  float* xln = w; w += TD;
  int*   eidx = (int*)w;

  const dim3 blk(256);

  k_init_loss<<<dim3(1), dim3(1), 0, stream>>>(loss);
  k_embed<<<dim3(TSEQ), blk, 0, stream>>>(idx, tok, pos, x);

  // QKV projections: [T,D] x [D,D] + bias
  const dim3 gqkv((DMODEL + 63) / 64, (TSEQ + 31) / 32, 1);
  k_wmma_gemm<false,false,true,false,false><<<gqkv, blk, 0, stream>>>(
      x, DMODEL, 0, nullptr, 0, Wq, DMODEL, 0, bq, 0,
      q, DMODEL, 0, nullptr, nullptr, TSEQ, DMODEL, DMODEL, 1.0f);
  k_wmma_gemm<false,false,true,false,false><<<gqkv, blk, 0, stream>>>(
      x, DMODEL, 0, nullptr, 0, Wk, DMODEL, 0, bk, 0,
      kk, DMODEL, 0, nullptr, nullptr, TSEQ, DMODEL, DMODEL, 1.0f);
  k_wmma_gemm<false,false,true,false,false><<<gqkv, blk, 0, stream>>>(
      x, DMODEL, 0, nullptr, 0, Wv, DMODEL, 0, bv, 0,
      v, DMODEL, 0, nullptr, nullptr, TSEQ, DMODEL, DMODEL, 1.0f);

  k_rotary<<<dim3(TSEQ), blk, 0, stream>>>(q, kk, qr, kr);

  // scores = qr @ kr^T / sqrt(D)   (B transposed: kr is [T,D])
  const dim3 gsc((TSEQ + 63) / 64, (TSEQ + 31) / 32, 1);
  k_wmma_gemm<false,true,false,false,false><<<gsc, blk, 0, stream>>>(
      qr, DMODEL, 0, nullptr, 0, kr, DMODEL, 0, nullptr, 0,
      sc, TSEQ, 0, nullptr, nullptr, TSEQ, TSEQ, DMODEL,
      0.044194173824159216f /* 1/sqrt(512) */);

  k_topk_softmax<<<dim3(TSEQ), blk, 0, stream>>>(sc);

  // attention out: w @ v  -> xf
  k_wmma_gemm<false,false,false,false,false><<<gqkv, blk, 0, stream>>>(
      sc, TSEQ, 0, nullptr, 0, v, DMODEL, 0, nullptr, 0,
      xf, DMODEL, 0, nullptr, nullptr, TSEQ, DMODEL, TSEQ, 1.0f);

  // MoE layers
  for (int l = 0; l < NLAYER; ++l) {
    const float* gWl = gateW + (long long)l * DMODEL * NEXP;
    const float* W1l = W1 + (long long)l * NEXP * DMODEL * HID;
    const float* b1l = b1 + (long long)l * NEXP * HID;
    const float* W2l = W2 + (long long)l * NEXP * HID * DMODEL;
    const float* b2l = b2 + (long long)l * NEXP * DMODEL;

    const dim3 gg((NEXP + 63) / 64, (TSEQ + 31) / 32, 1);
    k_wmma_gemm<false,false,false,false,false><<<gg, blk, 0, stream>>>(
        xf, DMODEL, 0, nullptr, 0, gWl, NEXP, 0, nullptr, 0,
        gl, NEXP, 0, nullptr, nullptr, TSEQ, NEXP, DMODEL, 1.0f);

    k_gate_softmax<<<dim3(TSEQ), dim3(32), 0, stream>>>(gl, pr);
    k_expert_topc<<<dim3(NEXP), blk, 0, stream>>>(pr, g, eidx, mp);
    k_aux<<<dim3(1), dim3(1), 0, stream>>>(mp, loss);

    // h[e] = gelu(gather(xf, eidx[e]) @ W1[e] + b1[e])   (batched over experts)
    const dim3 g1((HID + 63) / 64, (CAP + 31) / 32, NEXP);
    k_wmma_gemm<true,false,true,true,false><<<g1, blk, 0, stream>>>(
        xf, DMODEL, 0, eidx, CAP,
        W1l, HID, (long long)DMODEL * HID, b1l, HID,
        h, HID, (long long)CAP * HID, nullptr, nullptr,
        CAP, HID, DMODEL, 1.0f);

    // xf[tok] += (h[e] @ W2[e] + b2[e]) * g[e,c]   (scatter atomic combine)
    const dim3 g2((DMODEL + 63) / 64, (CAP + 31) / 32, NEXP);
    k_wmma_gemm<false,false,true,false,true><<<g2, blk, 0, stream>>>(
        h, HID, (long long)CAP * HID, nullptr, CAP,
        W2l, DMODEL, (long long)HID * DMODEL, b2l, DMODEL,
        xf, DMODEL, 0, eidx, g,
        CAP, DMODEL, HID, 1.0f);
  }

  k_layernorm<<<dim3(TSEQ), blk, 0, stream>>>(xf, lng, lnb, xln);

  // logits = xln @ head_W^T   (head_W is [V,D] -> BT layout, ldb=D)
  const dim3 gh((VOCAB + 63) / 64, (TSEQ + 31) / 32, 1);
  k_wmma_gemm<false,true,false,false,false><<<gh, blk, 0, stream>>>(
      xln, DMODEL, 0, nullptr, 0, headW, DMODEL, 0, nullptr, 0,
      logits, VOCAB, 0, nullptr, nullptr, TSEQ, VOCAB, DMODEL, 1.0f);
}
